// CVRPGNNBase_40398462386331
// MI455X (gfx1250) — matrix operand order repaired and checked by reading, not compiled, not run on hardware
//
#include <hip/hip_runtime.h>

#define NN 10000
#define NE 320000
#define NG 64
#define ESL (NE + NN)   // edges + self loops

typedef __attribute__((ext_vector_type(16))) _Float16 v16h;
typedef __attribute__((ext_vector_type(8)))  _Float16 v8h;
typedef __attribute__((ext_vector_type(8)))  float    v8f;

// ---- ordered-uint encoding for float atomicMax ----
__device__ __forceinline__ unsigned fenc(float f) {
  unsigned u = __float_as_uint(f);
  return (u & 0x80000000u) ? ~u : (u | 0x80000000u);
}
__device__ __forceinline__ float fdec(unsigned k) {
  return (k & 0x80000000u) ? __uint_as_float(k & 0x7FFFFFFFu)
                           : __uint_as_float(~k);
}
#define ENC_NEG_INF 0x007FFFFFu  // fenc(-inf)

// ---------------------------------------------------------------- packing
// Repack W[K,Nout] (f32, row-major) into WMMA B-fragment-major f16:
// P[((kt*(Nout/16)+nt)*32 + lane)*16 + kk] = W[(kt*32 + (kk&8)*2 + (lane>=16)*8 + (kk&7))*Nout
//                                              + nt*16 + (lane&15)]
// so each lane's v16h fragment is one contiguous 32-byte load.
__global__ void pack_wfrag(const float* __restrict__ W, _Float16* __restrict__ P,
                           int K, int Nout) {
  int idx = blockIdx.x * blockDim.x + threadIdx.x;
  int total = K * Nout;
  if (idx >= total) return;
  int kk = idx & 15;
  int L = (idx >> 4) & 31;
  int tile = idx >> 9;
  int ntiles = Nout >> 4;
  int nt = tile % ntiles;
  int kt = tile / ntiles;
  int krow = kt * 32 + ((kk & 8) << 1) + ((L >> 4) << 3) + (kk & 7);
  int col = nt * 16 + (L & 15);
  P[idx] = (_Float16)W[(size_t)krow * Nout + col];
}

// ---------------------------------------------------------- node encoder
__global__ void node_encode(const float* __restrict__ x, const float* __restrict__ W,
                            const float* __restrict__ b, const float* __restrict__ g,
                            const float* __restrict__ beta, float* __restrict__ out) {
  int lane = threadIdx.x & 31;
  int row  = blockIdx.x * (blockDim.x >> 5) + (threadIdx.x >> 5);
  if (row >= NN) return;
  const float* xr = x + row * 8;
  float xv[8];
#pragma unroll
  for (int k = 0; k < 8; ++k) xv[k] = xr[k];
  float y0 = b[lane], y1 = b[lane + 32];
#pragma unroll
  for (int k = 0; k < 8; ++k) {
    y0 += xv[k] * W[k * 64 + lane];
    y1 += xv[k] * W[k * 64 + lane + 32];
  }
  y0 = fmaxf(y0, 0.f); y1 = fmaxf(y1, 0.f);
  float s1 = y0 + y1, s2 = y0 * y0 + y1 * y1;
#pragma unroll
  for (int off = 16; off > 0; off >>= 1) {
    s1 += __shfl_xor(s1, off);
    s2 += __shfl_xor(s2, off);
  }
  float m = s1 * (1.f / 64.f);
  float var = s2 * (1.f / 64.f) - m * m;
  float r = rsqrtf(var + 1e-5f);
  out[row * 64 + lane]      = (y0 - m) * r * g[lane] + beta[lane];
  out[row * 64 + lane + 32] = (y1 - m) * r * g[lane + 32] + beta[lane + 32];
}

// ---------------------------------------------------------- edge encoder
__global__ void edge_encode(const float* __restrict__ ea, const float* __restrict__ W,
                            const float* __restrict__ b, const float* __restrict__ g,
                            const float* __restrict__ beta, float* __restrict__ out) {
  int lane = threadIdx.x & 31;
  int e = blockIdx.x * (blockDim.x >> 5) + (threadIdx.x >> 5);
  if (e >= NE) return;
  const float* ar = ea + e * 4;
  float y = b[lane];
#pragma unroll
  for (int k = 0; k < 4; ++k) y += ar[k] * W[k * 32 + lane];
  y = fmaxf(y, 0.f);
  float s1 = y, s2 = y * y;
#pragma unroll
  for (int off = 16; off > 0; off >>= 1) {
    s1 += __shfl_xor(s1, off);
    s2 += __shfl_xor(s2, off);
  }
  float m = s1 * (1.f / 32.f);
  float var = s2 * (1.f / 32.f) - m * m;
  float r = rsqrtf(var + 1e-5f);
  out[e * 32 + lane] = (y - m) * r * g[lane] + beta[lane];
}

// --------------------------------------------------- A fragment helpers
__device__ __forceinline__ v16h a_frag_global(const float* xrow, int k0, int akb) {
  const float4* p0 = (const float4*)(xrow + k0 + akb);
  const float4* p1 = (const float4*)(xrow + k0 + 16 + akb);
  float4 a0 = p0[0], a1 = p0[1], b0 = p1[0], b1 = p1[1];
  v16h a;
  a[0] = (_Float16)a0.x;  a[1] = (_Float16)a0.y;
  a[2] = (_Float16)a0.z;  a[3] = (_Float16)a0.w;
  a[4] = (_Float16)a1.x;  a[5] = (_Float16)a1.y;
  a[6] = (_Float16)a1.z;  a[7] = (_Float16)a1.w;
  a[8] = (_Float16)b0.x;  a[9] = (_Float16)b0.y;
  a[10] = (_Float16)b0.z; a[11] = (_Float16)b0.w;
  a[12] = (_Float16)b1.x; a[13] = (_Float16)b1.y;
  a[14] = (_Float16)b1.z; a[15] = (_Float16)b1.w;
  return a;
}
__device__ __forceinline__ v16h a_frag_lds(const _Float16* base, int k0, int akb) {
  v8h lo = *(const v8h*)(base + k0 + akb);
  v8h hv = *(const v8h*)(base + k0 + 16 + akb);
  return __builtin_shufflevector(lo, hv, 0, 1, 2, 3, 4, 5, 6, 7,
                                 8, 9, 10, 11, 12, 13, 14, 15);
}

// ------------------------------------------------- WMMA GEMM  Y = X @ W
// X:[M,K] f32 row-major, Wp: fragment-packed f16, Y:[M,Nout] f32.
// One wave per block computes a 16x64 tile. grid = (M/16, Nout/64).
__global__ __launch_bounds__(32) void gemm_xw(const float* __restrict__ X,
                                              const v16h* __restrict__ Wp,
                                              float* __restrict__ Y, int K, int Nout) {
  const int lane = threadIdx.x;
  const int hi = lane >> 4, lr = lane & 15;
  const int m0 = blockIdx.x * 16;
  const int n0 = blockIdx.y * 64;
  const int akb = hi ? 8 : 0;
  const int ntiles = Nout >> 4;
  const int nt0 = n0 >> 4;
  v8f acc[4] = {{}, {}, {}, {}};
  const float* xrow = X + (size_t)(m0 + lr) * K;
  for (int k0 = 0; k0 < K; k0 += 32) {
    v16h a = a_frag_global(xrow, k0, akb);
    const int kt = k0 >> 5;
#pragma unroll
    for (int nt = 0; nt < 4; ++nt) {
      v16h bf = Wp[((size_t)kt * ntiles + nt0 + nt) * 32 + lane];
      acc[nt] = __builtin_amdgcn_wmma_f32_16x16x32_f16(false, a, false, bf,
                                                       (short)0, acc[nt], false, false);
    }
  }
#pragma unroll
  for (int nt = 0; nt < 4; ++nt)
#pragma unroll
    for (int v = 0; v < 8; ++v)
      Y[(size_t)(m0 + v + hi * 8) * Nout + n0 + nt * 16 + lr] = acc[nt][v];
}

// ---------------------------------------- attention coefficient dots
__global__ void att_dots(const float* __restrict__ h, const float* __restrict__ as_w,
                         const float* __restrict__ ad_w, float* __restrict__ asb,
                         float* __restrict__ adb, int total, int heads) {
  int idx = blockIdx.x * blockDim.x + threadIdx.x;
  if (idx >= total) return;
  int hh = idx % heads;
  const float* hp = h + (size_t)idx * 64;
  float ss = 0.f, sd = 0.f;
#pragma unroll 8
  for (int c = 0; c < 64; ++c) {
    float v = hp[c];
    ss += v * as_w[hh * 64 + c];
    sd += v * ad_w[hh * 64 + c];
  }
  asb[idx] = ss;
  adb[idx] = sd;
}

__global__ void init_nm(unsigned* __restrict__ nmax, float* __restrict__ nsum, int n) {
  int i = blockIdx.x * blockDim.x + threadIdx.x;
  if (i < n) { nmax[i] = ENC_NEG_INF; nsum[i] = 0.f; }
}

__global__ void edge_score(const float* __restrict__ asb, const float* __restrict__ adb,
                           const int* __restrict__ ei, float* __restrict__ esc,
                           unsigned* __restrict__ nmax, int heads) {
  int idx = blockIdx.x * blockDim.x + threadIdx.x;
  int tot = ESL * heads;
  if (idx >= tot) return;
  int e = idx / heads, hh = idx - e * heads;
  int s, d;
  if (e < NE) { s = ei[e]; d = ei[NE + e]; } else { s = d = e - NE; }
  float v = asb[s * heads + hh] + adb[d * heads + hh];
  v = v > 0.f ? v : 0.2f * v;
  esc[idx] = v;
  atomicMax(nmax + d * heads + hh, fenc(v));
}

__global__ void edge_exp(const int* __restrict__ ei, float* __restrict__ esc,
                         const unsigned* __restrict__ nmax, float* __restrict__ nsum,
                         int heads) {
  int idx = blockIdx.x * blockDim.x + threadIdx.x;
  int tot = ESL * heads;
  if (idx >= tot) return;
  int e = idx / heads, hh = idx - e * heads;
  int d = (e < NE) ? ei[NE + e] : (e - NE);
  float ex = expf(esc[idx] - fdec(nmax[d * heads + hh]));
  esc[idx] = ex;
  atomicAdd(nsum + d * heads + hh, ex);
}

__global__ void init_bias(float* __restrict__ out, const float* __restrict__ b,
                          int HC, int n) {
  int i = blockIdx.x * blockDim.x + threadIdx.x;
  if (i < n) out[i] = b[i % HC];
}

__global__ void gat_scatter(const float* __restrict__ hbuf, const float* __restrict__ esc,
                            const float* __restrict__ nsum, const int* __restrict__ ei,
                            float* __restrict__ out, int heads) {
  int e = blockIdx.x;
  int t = threadIdx.x;
  int hh = t >> 6;
  int s, d;
  if (e < NE) { s = ei[e]; d = ei[NE + e]; } else { s = d = e - NE; }
  int HC = heads * 64;
  float w = esc[e * heads + hh] / nsum[d * heads + hh];
  atomicAdd(out + (size_t)d * HC + t, hbuf[(size_t)s * HC + t] * w);
}

__global__ void elu_ip(float* __restrict__ x, int n) {
  int i = blockIdx.x * blockDim.x + threadIdx.x;
  if (i < n) { float v = x[i]; x[i] = v > 0.f ? v : (expf(v) - 1.f); }
}

// ----------------------------------------- fused edge classifier (WMMA)
__global__ __launch_bounds__(32) void edge_cls(const float* __restrict__ xn,
    const float* __restrict__ ef, const int* __restrict__ ei,
    const v16h* __restrict__ W1, const float* __restrict__ b1,
    const v16h* __restrict__ W2, const float* __restrict__ b2,
    const float* __restrict__ w3, const float* __restrict__ b3,
    float* __restrict__ pred) {
  __shared__ _Float16 lc[16 * 160];
  __shared__ _Float16 lh1[16 * 64];
  __shared__ float    lh2[16 * 64];
  const int lane = threadIdx.x;
  const int e0 = blockIdx.x * 16;
  for (int idx = lane; idx < 16 * 160; idx += 32) {
    int r = idx / 160, c = idx - r * 160;
    int e = e0 + r;
    float v;
    if (c < 64)       v = xn[(size_t)ei[e] * 64 + c];
    else if (c < 128) v = xn[(size_t)ei[NE + e] * 64 + (c - 64)];
    else              v = ef[(size_t)e * 32 + (c - 128)];
    lc[idx] = (_Float16)v;
  }
  __syncthreads();
  const int hi = lane >> 4, lr = lane & 15;
  const int akb = hi ? 8 : 0;
  v8f acc[4] = {{}, {}, {}, {}};
  for (int kt = 0; kt < 5; ++kt) {
    v16h a = a_frag_lds(lc + lr * 160, kt * 32, akb);
#pragma unroll
    for (int nt = 0; nt < 4; ++nt) {
      v16h bf = W1[((size_t)kt * 4 + nt) * 32 + lane];
      acc[nt] = __builtin_amdgcn_wmma_f32_16x16x32_f16(false, a, false, bf,
                                                       (short)0, acc[nt], false, false);
    }
  }
#pragma unroll
  for (int nt = 0; nt < 4; ++nt)
#pragma unroll
    for (int v = 0; v < 8; ++v) {
      int row = v + hi * 8, col = nt * 16 + lr;
      float t = acc[nt][v] + b1[col];
      lh1[row * 64 + col] = (_Float16)(t > 0.f ? t : 0.f);
    }
  __syncthreads();
  v8f acc2[4] = {{}, {}, {}, {}};
  for (int kt = 0; kt < 2; ++kt) {
    v16h a = a_frag_lds(lh1 + lr * 64, kt * 32, akb);
#pragma unroll
    for (int nt = 0; nt < 4; ++nt) {
      v16h bf = W2[((size_t)kt * 4 + nt) * 32 + lane];
      acc2[nt] = __builtin_amdgcn_wmma_f32_16x16x32_f16(false, a, false, bf,
                                                        (short)0, acc2[nt], false, false);
    }
  }
#pragma unroll
  for (int nt = 0; nt < 4; ++nt)
#pragma unroll
    for (int v = 0; v < 8; ++v) {
      int row = v + hi * 8, col = nt * 16 + lr;
      float t = acc2[nt][v] + b2[col];
      lh2[row * 64 + col] = t > 0.f ? t : 0.f;
    }
  __syncthreads();
  if (lane < 16) {
    float s = b3[0];
#pragma unroll 8
    for (int c = 0; c < 64; ++c) s += lh2[lane * 64 + c] * w3[c];
    pred[e0 + lane] = s;
  }
}

// ----------------------------------------------- fused node scorer (WMMA)
__global__ __launch_bounds__(32) void node_score(const float* __restrict__ xn,
    const v16h* __restrict__ W1, const float* __restrict__ b1,
    const float* __restrict__ w2, const float* __restrict__ b2,
    float* __restrict__ pred) {
  __shared__ float lh[16 * 64];
  const int lane = threadIdx.x;
  const int hi = lane >> 4, lr = lane & 15;
  const int akb = hi ? 8 : 0;
  const int n0 = blockIdx.x * 16;
  v8f acc[4] = {{}, {}, {}, {}};
  const float* xrow = xn + (size_t)(n0 + lr) * 64;
  for (int kt = 0; kt < 2; ++kt) {
    v16h a = a_frag_global(xrow, kt * 32, akb);
#pragma unroll
    for (int nt = 0; nt < 4; ++nt) {
      v16h bf = W1[((size_t)kt * 4 + nt) * 32 + lane];
      acc[nt] = __builtin_amdgcn_wmma_f32_16x16x32_f16(false, a, false, bf,
                                                       (short)0, acc[nt], false, false);
    }
  }
#pragma unroll
  for (int nt = 0; nt < 4; ++nt)
#pragma unroll
    for (int v = 0; v < 8; ++v) {
      int row = v + hi * 8, col = nt * 16 + lr;
      float t = acc[nt][v] + b1[col];
      lh[row * 64 + col] = t > 0.f ? t : 0.f;
    }
  __syncthreads();
  if (lane < 16) {
    float s = b2[0];
#pragma unroll 8
    for (int c = 0; c < 64; ++c) s += lh[lane * 64 + c] * w2[c];
    pred[n0 + lane] = s;
  }
}

// ------------------------------------------------------- graph pooling
__global__ void pool_init(float* __restrict__ gsum, unsigned* __restrict__ gmax,
                          float* __restrict__ gcnt) {
  int i = blockIdx.x * blockDim.x + threadIdx.x;
  if (i < NG * 64) { gsum[i] = 0.f; gmax[i] = ENC_NEG_INF; }
  if (i < NG) gcnt[i] = 0.f;
}
__global__ void pool_acc(const float* __restrict__ xn, const int* __restrict__ batch,
                         float* __restrict__ gsum, unsigned* __restrict__ gmax,
                         float* __restrict__ gcnt) {
  int i = blockIdx.x * blockDim.x + threadIdx.x;
  if (i >= NN * 64) return;
  int n = i >> 6, c = i & 63;
  int g = batch[n];
  float v = xn[i];
  atomicAdd(gsum + g * 64 + c, v);
  atomicMax(gmax + g * 64 + c, fenc(v));
  if (c == 0) atomicAdd(gcnt + g, 1.f);
}
__global__ void pool_fin(const float* __restrict__ gsum, const unsigned* __restrict__ gmax,
                         const float* __restrict__ gcnt, float* __restrict__ out) {
  int i = blockIdx.x * blockDim.x + threadIdx.x;
  if (i >= NG * 128) return;
  int g = i >> 7, j = i & 127;
  out[i] = (j < 64) ? gsum[g * 64 + j] / fmaxf(gcnt[g], 1.f)
                    : fdec(gmax[g * 64 + (j - 64)]);
}

// ===================================================================
extern "C" void kernel_launch(void* const* d_in, const int* in_sizes, int n_in,
                              void* d_out, int out_size, void* d_ws, size_t ws_size,
                              hipStream_t stream) {
  (void)in_sizes; (void)n_in; (void)out_size; (void)ws_size;
  const float* x     = (const float*)d_in[0];
  const float* ea    = (const float*)d_in[1];
  const int*   ei    = (const int*)d_in[2];
  const int*   batch = (const int*)d_in[3];
  const float* ne_w = (const float*)d_in[4],  * ne_b = (const float*)d_in[5];
  const float* ne_g = (const float*)d_in[6],  * ne_be = (const float*)d_in[7];
  const float* ee_w = (const float*)d_in[8],  * ee_b = (const float*)d_in[9];
  const float* ee_g = (const float*)d_in[10], * ee_be = (const float*)d_in[11];
  const float* g0_w = (const float*)d_in[12], * g0_as = (const float*)d_in[13];
  const float* g0_ad = (const float*)d_in[14], * g0_b = (const float*)d_in[15];
  const float* g1_w = (const float*)d_in[16], * g1_as = (const float*)d_in[17];
  const float* g1_ad = (const float*)d_in[18], * g1_b = (const float*)d_in[19];
  const float* g2_w = (const float*)d_in[20], * g2_as = (const float*)d_in[21];
  const float* g2_ad = (const float*)d_in[22], * g2_b = (const float*)d_in[23];
  const float* ec1_w = (const float*)d_in[24], * ec1_b = (const float*)d_in[25];
  const float* ec2_w = (const float*)d_in[26], * ec2_b = (const float*)d_in[27];
  const float* ec3_w = (const float*)d_in[28], * ec3_b = (const float*)d_in[29];
  const float* ns1_w = (const float*)d_in[30], * ns1_b = (const float*)d_in[31];
  const float* ns2_w = (const float*)d_in[32], * ns2_b = (const float*)d_in[33];

  // ---- workspace carve-out ----
  char* ws = (char*)d_ws;
  size_t off = 0;
  auto alloc = [&](size_t bytes) -> void* {
    void* p = ws + off;
    off = (off + bytes + 255) & ~(size_t)255;
    return p;
  };
  _Float16* g0w16  = (_Float16*)alloc(64 * 256 * 2);
  _Float16* g1w16  = (_Float16*)alloc(256 * 256 * 2);
  _Float16* g2w16  = (_Float16*)alloc(256 * 64 * 2);
  _Float16* ec1w16 = (_Float16*)alloc(160 * 64 * 2);
  _Float16* ec2w16 = (_Float16*)alloc(64 * 64 * 2);
  _Float16* ns1w16 = (_Float16*)alloc(64 * 64 * 2);
  float* xn0 = (float*)alloc((size_t)NN * 64 * 4);
  float* H   = (float*)alloc((size_t)NN * 256 * 4);
  float* O1  = (float*)alloc((size_t)NN * 256 * 4);
  float* O2  = (float*)alloc((size_t)NN * 256 * 4);
  float* efb = (float*)alloc((size_t)NE * 32 * 4);
  float* asb = (float*)alloc((size_t)NN * 4 * 4);
  float* adb = (float*)alloc((size_t)NN * 4 * 4);
  float* esc = (float*)alloc((size_t)ESL * 4 * 4);
  unsigned* nmax = (unsigned*)alloc((size_t)NN * 4 * 4);
  float* nsum = (float*)alloc((size_t)NN * 4 * 4);
  float* gsum = (float*)alloc((size_t)NG * 64 * 4);
  unsigned* gmax = (unsigned*)alloc((size_t)NG * 64 * 4);
  float* gcnt = (float*)alloc((size_t)NG * 4);

  float* out = (float*)d_out;
  float* edge_pred  = out;
  float* node_pred  = out + NE;
  float* graph_feat = out + NE + NN;
  float* xn_final   = out + NE + NN + NG * 128;

  // ---- weight repack to fragment-major f16 ----
  auto pack = [&](const float* s, _Float16* d, int K, int Nout) {
    int n = K * Nout;
    pack_wfrag<<<(n + 255) / 256, 256, 0, stream>>>(s, d, K, Nout);
  };
  pack(g0_w, g0w16, 64, 256);
  pack(g1_w, g1w16, 256, 256);
  pack(g2_w, g2w16, 256, 64);
  pack(ec1_w, ec1w16, 160, 64);
  pack(ec2_w, ec2w16, 64, 64);
  pack(ns1_w, ns1w16, 64, 64);

  // ---- encoders ----
  node_encode<<<(NN + 7) / 8, 256, 0, stream>>>(x, ne_w, ne_b, ne_g, ne_be, xn0);
  edge_encode<<<NE / 8, 256, 0, stream>>>(ea, ee_w, ee_b, ee_g, ee_be, efb);

  // ---- GAT layers ----
  auto run_gat = [&](const float* Xin, int K, const _Float16* Wh, const float* bgat,
                     const float* as_w, const float* ad_w, int heads, float* Yout,
                     bool do_elu) {
    int HC = heads * 64;
    dim3 gg(NN / 16, HC / 64);
    gemm_xw<<<gg, 32, 0, stream>>>(Xin, (const v16h*)Wh, H, K, HC);
    int natt = NN * heads;
    att_dots<<<(natt + 255) / 256, 256, 0, stream>>>(H, as_w, ad_w, asb, adb, natt, heads);
    init_nm<<<(natt + 255) / 256, 256, 0, stream>>>(nmax, nsum, natt);
    int tot = ESL * heads;
    edge_score<<<(tot + 255) / 256, 256, 0, stream>>>(asb, adb, ei, esc, nmax, heads);
    edge_exp<<<(tot + 255) / 256, 256, 0, stream>>>(ei, esc, nmax, nsum, heads);
    int nout = NN * HC;
    init_bias<<<(nout + 255) / 256, 256, 0, stream>>>(Yout, bgat, HC, nout);
    gat_scatter<<<ESL, HC, 0, stream>>>(H, esc, nsum, ei, Yout, heads);
    if (do_elu) elu_ip<<<(nout + 255) / 256, 256, 0, stream>>>(Yout, nout);
  };
  run_gat(xn0, 64,  g0w16, g0_b, g0_as, g0_ad, 4, O1, true);
  run_gat(O1, 256,  g1w16, g1_b, g1_as, g1_ad, 4, O2, true);
  run_gat(O2, 256,  g2w16, g2_b, g2_as, g2_ad, 1, xn_final, false);

  // ---- heads ----
  edge_cls<<<NE / 16, 32, 0, stream>>>(xn_final, efb, ei, (const v16h*)ec1w16, ec1_b,
                                       (const v16h*)ec2w16, ec2_b, ec3_w, ec3_b,
                                       edge_pred);
  node_score<<<NN / 16, 32, 0, stream>>>(xn_final, (const v16h*)ns1w16, ns1_b,
                                         ns2_w, ns2_b, node_pred);

  // ---- graph pooling ----
  pool_init<<<(NG * 64 + 255) / 256, 256, 0, stream>>>(gsum, gmax, gcnt);
  pool_acc<<<(NN * 64 + 255) / 256, 256, 0, stream>>>(xn_final, batch, gsum, gmax, gcnt);
  pool_fin<<<(NG * 128 + 255) / 256, 256, 0, stream>>>(gsum, gmax, gcnt, graph_feat);
}
